// SlotEncoder_37855841747710
// MI455X (gfx1250) — matrix operand order repaired and checked by reading, not compiled, not run on hardware
//
#include <hip/hip_runtime.h>
#include <hip/hip_bf16.h>

typedef __attribute__((ext_vector_type(16))) _Float16 v16h;
typedef __attribute__((ext_vector_type(8)))  float    v8f;

#define NB   512
#define NPIX 900
#define NPAD 928     // 900 padded to multiple of 32 for WMMA K / 16 for tiles
#define DDIM 128
#define NS   10
#define SPAD 16      // slots padded to 16 rows
#define GH   30
#define GW   30

// ---------------------------------------------------------------------------
// WMMA fragment loaders (CDNA5 16x16x32 f16 layouts, ISA 7.12.2)
// ---------------------------------------------------------------------------

// A: 16x32 f16 tile, row-major source with row stride lda.
// lanes 0-15: M=lane, K 0-7 (halves 0-7) and K 16-23 (halves 8-15)
// lanes 16-31: M=lane-16, K 8-15 and K 24-31
__device__ inline v16h frag_a(const _Float16* A, int lda, int lane) {
  int m  = lane & 15;
  int kb = (lane & 16) ? 8 : 0;
  const _Float16* p = A + (size_t)m * lda + kb;
  v16h r;
#pragma unroll
  for (int i = 0; i < 8; i++) r[i] = p[i];
#pragma unroll
  for (int i = 0; i < 8; i++) r[8 + i] = p[16 + i];
  return r;
}

// Same A fragment but source is f32 (attention weights), converted on load.
__device__ inline v16h frag_a_f32(const float* A, int lda, int lane) {
  int m  = lane & 15;
  int kb = (lane & 16) ? 8 : 0;
  const float* p = A + (size_t)m * lda + kb;
  v16h r;
#pragma unroll
  for (int i = 0; i < 8; i++) r[i] = (_Float16)p[i];
#pragma unroll
  for (int i = 0; i < 8; i++) r[8 + i] = (_Float16)p[16 + i];
  return r;
}

// B: 32x16 f16 tile given as B-transposed (N x K row-major, stride ldbt).
// lanes 0-15: N=lane, K 0-7 and K 16-23 ; lanes 16-31: N=lane-16, K 8-15 and K 24-31.
// Two 16B loads per lane.
__device__ inline v16h frag_bt(const _Float16* Bt, int ldbt, int lane) {
  int n  = lane & 15;
  int kb = (lane & 16) ? 8 : 0;
  const _Float16* p = Bt + (size_t)n * ldbt + kb;
  v16h r;
#pragma unroll
  for (int i = 0; i < 8; i++) r[i] = p[i];
#pragma unroll
  for (int i = 0; i < 8; i++) r[8 + i] = p[16 + i];
  return r;
}

__device__ inline v8f wmma_f16(v16h a, v16h b, v8f c) {
  return __builtin_amdgcn_wmma_f32_16x16x32_f16(false, a, false, b, (short)0, c,
                                                false, false);
}

// ---------------------------------------------------------------------------
// Weight convert (optionally transpose) f32 -> f16, dst is [N,K] row-major
// ---------------------------------------------------------------------------
__global__ void wconv_kernel(const float* src, _Float16* dst, int N, int K, int trans) {
  int i = blockIdx.x * blockDim.x + threadIdx.x;
  if (i >= N * K) return;
  int n = i / K, k = i % K;
  float v = trans ? src[(size_t)k * N + n] : src[i];
  dst[i] = (_Float16)v;
}

// ---------------------------------------------------------------------------
// Embed + input LayerNorm -> x f16 [NB*NPAD, 128] (pad rows zeroed)
// one wave (32 lanes) per row, 4 channels per lane
// ---------------------------------------------------------------------------
__global__ void embed_ln_kernel(const int* grid, const float* cemb, const float* pemb,
                                const float* g, const float* bp, _Float16* x) {
  int row  = blockIdx.x * blockDim.y + threadIdx.y;
  int lane = threadIdx.x;
  if (row >= NB * NPAD) return;
  int b = row / NPAD, n = row % NPAD;
  _Float16* xr = x + (size_t)row * DDIM;
  if (n >= NPIX) {
#pragma unroll
    for (int i = 0; i < 4; i++) xr[lane * 4 + i] = (_Float16)0.f;
    return;
  }
  int c = grid[b * NPIX + n];
  float v[4]; float s = 0.f;
#pragma unroll
  for (int i = 0; i < 4; i++) {
    int d = lane * 4 + i;
    v[i] = cemb[c * DDIM + d] + pemb[n * DDIM + d];
    s += v[i];
  }
#pragma unroll
  for (int m = 16; m >= 1; m >>= 1) s += __shfl_xor(s, m, 32);
  float mu = s * (1.f / 128.f);
  float vs = 0.f;
#pragma unroll
  for (int i = 0; i < 4; i++) { float d = v[i] - mu; vs += d * d; }
#pragma unroll
  for (int m = 16; m >= 1; m >>= 1) vs += __shfl_xor(vs, m, 32);
  float rstd = rsqrtf(vs * (1.f / 128.f) + 1e-5f);
#pragma unroll
  for (int i = 0; i < 4; i++) {
    int d = lane * 4 + i;
    xr[d] = (_Float16)((v[i] - mu) * rstd * g[d] + bp[d]);
  }
}

// ---------------------------------------------------------------------------
// Slot init: slots = mu + sigma*noise, padded to 16 rows (pad = 0)
// ---------------------------------------------------------------------------
__global__ void slots_init_kernel(const float* noise, const float* mu, const float* sg,
                                  float* slots, _Float16* slots_h) {
  int idx = blockIdx.x * blockDim.x + threadIdx.x;
  if (idx >= NB * SPAD * DDIM) return;
  int d = idx % DDIM; int s = (idx / DDIM) % SPAD; int b = idx / (DDIM * SPAD);
  float v = 0.f;
  if (s < NS) v = mu[d] + sg[d] * noise[((size_t)(b * NS + s)) * DDIM + d];
  slots[idx]   = v;
  slots_h[idx] = (_Float16)v;
}

// ---------------------------------------------------------------------------
// Slot LayerNorm -> f16 (pad rows zeroed), one wave per row
// ---------------------------------------------------------------------------
__global__ void slot_ln_kernel(const float* slots, const float* g, const float* bp,
                               _Float16* out) {
  int row  = blockIdx.x * blockDim.y + threadIdx.y;
  int lane = threadIdx.x;
  if (row >= NB * SPAD) return;
  int s = row % SPAD;
  _Float16* o = out + (size_t)row * DDIM;
  if (s >= NS) {
#pragma unroll
    for (int i = 0; i < 4; i++) o[lane * 4 + i] = (_Float16)0.f;
    return;
  }
  const float* sr = slots + (size_t)row * DDIM;
  float v[4], sum = 0.f;
#pragma unroll
  for (int i = 0; i < 4; i++) { v[i] = sr[lane * 4 + i]; sum += v[i]; }
#pragma unroll
  for (int m = 16; m >= 1; m >>= 1) sum += __shfl_xor(sum, m, 32);
  float mu = sum * (1.f / 128.f), vs = 0.f;
#pragma unroll
  for (int i = 0; i < 4; i++) { float d = v[i] - mu; vs += d * d; }
#pragma unroll
  for (int m = 16; m >= 1; m >>= 1) vs += __shfl_xor(vs, m, 32);
  float r = rsqrtf(vs * (1.f / 128.f) + 1e-5f);
#pragma unroll
  for (int i = 0; i < 4; i++) {
    int d = lane * 4 + i;
    o[d] = (_Float16)((v[i] - mu) * r * g[d] + bp[d]);
  }
}

// ---------------------------------------------------------------------------
// Generic 16x16-tile WMMA GEMM: C[M,N] = A[M,K] (f16, row-major) x Bt[N,K] (f16)
// + optional bias, optional exact-GELU; writes f32 and/or f16 outputs.
// One wave per block; grid = (M/16, N/16).
// ---------------------------------------------------------------------------
__global__ void gemm16_kernel(const _Float16* A, int lda, const _Float16* Bt, int ldbt,
                              float* Cf, _Float16* Ch, int ldc, int K,
                              const float* bias, int act) {
  int m0 = blockIdx.x * 16, n0 = blockIdx.y * 16;
  int lane = threadIdx.x;
  v8f acc = {};
  for (int k0 = 0; k0 < K; k0 += 32) {
    v16h a = frag_a(A + (size_t)m0 * lda + k0, lda, lane);
    v16h b = frag_bt(Bt + (size_t)n0 * ldbt + k0, ldbt, lane);
    acc = wmma_f16(a, b, acc);
  }
  int n    = n0 + (lane & 15);
  int mofs = (lane & 16) ? 8 : 0;
  float bv = bias ? bias[n] : 0.f;
#pragma unroll
  for (int r = 0; r < 8; r++) {
    int m   = m0 + mofs + r;
    float v = acc[r] + bv;
    if (act == 1) v = 0.5f * v * (1.f + erff(v * 0.70710678f));  // exact GELU
    if (Cf) Cf[(size_t)m * ldc + n] = v;
    if (Ch) Ch[(size_t)m * ldc + n] = (_Float16)v;
  }
}

// ---------------------------------------------------------------------------
// Fused K/V projection with A-fragment reuse: one wave per 16-pixel M-tile.
// A (16x128) is loaded ONCE into 4 fragments, then all 16 output n-tiles
// (8 for K, 8 for V) are computed against the L2-resident weights.
// k stored row-major [NB*NPAD,128] (serves as Bt for logits); v stored
// transposed per-batch [NB][128][NPAD] (serves as Bt for updates).
// grid = MROWS/16 blocks, block 32.
// ---------------------------------------------------------------------------
__global__ void kv_gemm_kernel(const _Float16* X, const _Float16* Wkt, const _Float16* Wvt,
                               _Float16* Kbuf, _Float16* VT) {
  int m0   = blockIdx.x * 16;
  int lane = threadIdx.x;
  const _Float16* Arow = X + (size_t)m0 * DDIM;
  v16h a0 = frag_a(Arow +  0, DDIM, lane);
  v16h a1 = frag_a(Arow + 32, DDIM, lane);
  v16h a2 = frag_a(Arow + 64, DDIM, lane);
  v16h a3 = frag_a(Arow + 96, DDIM, lane);
  int bb   = m0 / NPAD;            // 16-row tile never crosses batch (928%16==0)
  int col0 = m0 % NPAD;
  int nlane = lane & 15;
  int mofs  = (lane & 16) ? 8 : 0;
#pragma unroll
  for (int nt = 0; nt < 16; nt++) {
    int n0 = (nt & 7) * 16;
    const _Float16* Bt = ((nt < 8) ? Wkt : Wvt) + (size_t)n0 * DDIM;
    v8f acc = {};
    acc = wmma_f16(a0, frag_bt(Bt +  0, DDIM, lane), acc);
    acc = wmma_f16(a1, frag_bt(Bt + 32, DDIM, lane), acc);
    acc = wmma_f16(a2, frag_bt(Bt + 64, DDIM, lane), acc);
    acc = wmma_f16(a3, frag_bt(Bt + 96, DDIM, lane), acc);
    int n = n0 + nlane;
    if (nt < 8) {
#pragma unroll
      for (int r = 0; r < 8; r++)
        Kbuf[(size_t)(m0 + mofs + r) * DDIM + n] = (_Float16)acc[r];
    } else {
      _Float16* vt = VT + (size_t)bb * DDIM * NPAD + (size_t)n * NPAD + col0 + mofs;
#pragma unroll
      for (int r = 0; r < 8; r++) vt[r] = (_Float16)acc[r];
    }
  }
}

// ---------------------------------------------------------------------------
// logits[b,s,n] = (q[b,s,:] . k[b,n,:]) / sqrt(D).
// q A-fragments loaded once per wave, 29 n-tiles per block.
// grid = (NB, 2), block 32.
// ---------------------------------------------------------------------------
__global__ void logits_kernel(const _Float16* Q, const _Float16* Kbuf, float* logits) {
  int b = blockIdx.x, half = blockIdx.y, lane = threadIdx.x;
  const _Float16* A = Q + (size_t)b * SPAD * DDIM;
  v16h a0 = frag_a(A +  0, DDIM, lane);
  v16h a1 = frag_a(A + 32, DDIM, lane);
  v16h a2 = frag_a(A + 64, DDIM, lane);
  v16h a3 = frag_a(A + 96, DDIM, lane);
  const float scale = 0.08838834764831845f;  // 1/sqrt(128)
  int nlane = lane & 15;
  int mofs  = (lane & 16) ? 8 : 0;
  float* L = logits + (size_t)b * SPAD * NPAD;
  int nt0 = half * 29;
  for (int nt = nt0; nt < nt0 + 29; nt++) {
    const _Float16* Bt = Kbuf + (size_t)b * NPAD * DDIM + (size_t)nt * 16 * DDIM;
    v8f acc = {};
    acc = wmma_f16(a0, frag_bt(Bt +  0, DDIM, lane), acc);
    acc = wmma_f16(a1, frag_bt(Bt + 32, DDIM, lane), acc);
    acc = wmma_f16(a2, frag_bt(Bt + 64, DDIM, lane), acc);
    acc = wmma_f16(a3, frag_bt(Bt + 96, DDIM, lane), acc);
    int n = nt * 16 + nlane;
#pragma unroll
    for (int r = 0; r < 8; r++) L[(size_t)(mofs + r) * NPAD + n] = acc[r] * scale;
  }
}

// ---------------------------------------------------------------------------
// Softmax over slots (axis=1) per key; pad slots/keys -> 0. In place.
// ---------------------------------------------------------------------------
__global__ void softmax_kernel(float* attn) {
  int b = blockIdx.x;
  int n = blockIdx.y * blockDim.x + threadIdx.x;
  if (n >= NPAD) return;
  float* A = attn + (size_t)b * SPAD * NPAD + n;
  if (n >= NPIX) {
    for (int s = 0; s < SPAD; s++) A[(size_t)s * NPAD] = 0.f;
    return;
  }
  float v[NS]; float mx = -1e30f;
#pragma unroll
  for (int s = 0; s < NS; s++) { v[s] = A[(size_t)s * NPAD]; mx = fmaxf(mx, v[s]); }
  float sum = 0.f;
#pragma unroll
  for (int s = 0; s < NS; s++) { v[s] = expf(v[s] - mx); sum += v[s]; }
  float inv = 1.f / sum;
#pragma unroll
  for (int s = 0; s < NS; s++) A[(size_t)s * NPAD] = v[s] * inv;
  for (int s = NS; s < SPAD; s++) A[(size_t)s * NPAD] = 0.f;
}

// ---------------------------------------------------------------------------
// Renormalize each slot's attention over keys. One block per (b,s) row.
// ---------------------------------------------------------------------------
__global__ void renorm_kernel(float* attn) {
  float* A = attn + (size_t)blockIdx.x * NPAD;
  __shared__ float red[256];
  float s = 0.f;
  for (int n = threadIdx.x; n < NPAD; n += 256) s += A[n];
  red[threadIdx.x] = s; __syncthreads();
  for (int st = 128; st > 0; st >>= 1) {
    if (threadIdx.x < st) red[threadIdx.x] += red[threadIdx.x + st];
    __syncthreads();
  }
  float inv = 1.f / (red[0] + 1e-8f);
  for (int n = threadIdx.x; n < NPAD; n += 256) A[n] *= inv;
}

// ---------------------------------------------------------------------------
// updates[b] = attn[b] (16x928, f32->f16 on load) @ v[b] (via vT as Bt).
// One block per batch: each attn A-fragment is loaded once per K-chunk and
// reused across all 8 feature n-tiles (8 independent accumulator chains).
// grid = NB, block 32.
// ---------------------------------------------------------------------------
__global__ void updates_kernel(const float* attn, const _Float16* VT, _Float16* Uh) {
  int b = blockIdx.x, lane = threadIdx.x;
  const float*    A = attn + (size_t)b * SPAD * NPAD;
  const _Float16* V = VT + (size_t)b * DDIM * NPAD;
  v8f acc[8];
  v8f z = {};
#pragma unroll
  for (int nt = 0; nt < 8; nt++) acc[nt] = z;
  for (int k0 = 0; k0 < NPAD; k0 += 32) {
    v16h a = frag_a_f32(A + k0, NPAD, lane);
#pragma unroll
    for (int nt = 0; nt < 8; nt++) {
      v16h bf = frag_bt(V + (size_t)nt * 16 * NPAD + k0, NPAD, lane);
      acc[nt] = wmma_f16(a, bf, acc[nt]);
    }
  }
  int nlane = lane & 15;
  int mofs  = (lane & 16) ? 8 : 0;
  _Float16* U = Uh + (size_t)b * SPAD * DDIM;
#pragma unroll
  for (int nt = 0; nt < 8; nt++) {
    int n = nt * 16 + nlane;
#pragma unroll
    for (int r = 0; r < 8; r++) U[(size_t)(mofs + r) * DDIM + n] = (_Float16)acc[nt][r];
  }
}

// ---------------------------------------------------------------------------
// GRU pointwise: r,z,n gates + state update (biases folded here)
// ---------------------------------------------------------------------------
__global__ void gru_kernel(const float* gi, const float* gh, const float* bih,
                           const float* bhh, float* slots, _Float16* slots_h) {
  int idx = blockIdx.x * blockDim.x + threadIdx.x;
  if (idx >= NB * NS * DDIM) return;
  int d = idx % DDIM; int s = (idx / DDIM) % NS; int b = idx / (DDIM * NS);
  size_t row = (size_t)(b * SPAD + s);
  const float* gir = gi + row * 384;
  const float* ghr = gh + row * 384;
  float r  = 1.f / (1.f + expf(-(gir[d] + bih[d] + ghr[d] + bhh[d])));
  float z  = 1.f / (1.f + expf(-(gir[128 + d] + bih[128 + d] + ghr[128 + d] + bhh[128 + d])));
  float hn = ghr[256 + d] + bhh[256 + d];
  float nn = tanhf(gir[256 + d] + bih[256 + d] + r * hn);
  float prev = slots[row * DDIM + d];
  float v = (1.f - z) * nn + z * prev;
  slots[row * DDIM + d]   = v;
  slots_h[row * DDIM + d] = (_Float16)v;
}

__global__ void mlp_add_kernel(const float* mlpout, float* slots, _Float16* slots_h) {
  int idx = blockIdx.x * blockDim.x + threadIdx.x;
  if (idx >= NB * NS * DDIM) return;
  int d = idx % DDIM; int s = (idx / DDIM) % NS; int b = idx / (DDIM * NS);
  size_t o = (size_t)(b * SPAD + s) * DDIM + d;
  float v = slots[o] + mlpout[o];
  slots[o]   = v;
  slots_h[o] = (_Float16)v;
}

// ---------------------------------------------------------------------------
// Output assembly
// ---------------------------------------------------------------------------
__global__ void out_slots_kernel(const float* slots, float* out) {
  int idx = blockIdx.x * blockDim.x + threadIdx.x;
  if (idx >= NB * NS * DDIM) return;
  int d = idx % DDIM; int s = (idx / DDIM) % NS; int b = idx / (DDIM * NS);
  out[idx] = slots[(size_t)(b * SPAD + s) * DDIM + d];
}

__global__ void out_attn_kernel(const float* attn, float* out) {
  int idx = blockIdx.x * blockDim.x + threadIdx.x;
  if (idx >= NB * NS * NPIX) return;
  int n = idx % NPIX; int s = (idx / NPIX) % NS; int b = idx / (NPIX * NS);
  out[idx] = attn[(size_t)b * SPAD * NPAD + (size_t)s * NPAD + n];
}

// props per (b,s): size, centroid, extent, color distribution
__global__ void props_kernel(const float* attn, const int* grid, float* out) {
  int b = blockIdx.x / NS, s = blockIdx.x % NS;
  const float* A = attn + (size_t)b * SPAD * NPAD + (size_t)s * NPAD;
  __shared__ float red[256];
  int t = threadIdx.x;
  auto reduce = [&](float v) -> float {
    red[t] = v; __syncthreads();
    for (int st = 128; st > 0; st >>= 1) {
      if (t < st) red[t] += red[t + st];
      __syncthreads();
    }
    float r = red[0]; __syncthreads();
    return r;
  };
  float s_raw = 0.f, s_a = 0.f, s_y = 0.f, s_x = 0.f;
  float cd[10];
#pragma unroll
  for (int c = 0; c < 10; c++) cd[c] = 0.f;
  for (int n = t; n < NPIX; n += 256) {
    float av = A[n];
    int h = n / GW, w = n % GW;
    float a = av + 1e-8f;
    s_raw += av; s_a += a;
    s_y += a * (h * (1.f / 29.f));
    s_x += a * (w * (1.f / 29.f));
    cd[grid[b * NPIX + n]] += av;
  }
  float sumraw = reduce(s_raw);
  float denom  = reduce(s_a);
  float cy = reduce(s_y) / denom;
  float cx = reduce(s_x) / denom;
  float vy_p = 0.f, vx_p = 0.f;
  for (int n = t; n < NPIX; n += 256) {
    float a = A[n] + 1e-8f;
    int h = n / GW, w = n % GW;
    float dy = h * (1.f / 29.f) - cy, dx = w * (1.f / 29.f) - cx;
    vy_p += a * dy * dy;
    vx_p += a * dx * dx;
  }
  float vy = reduce(vy_p) / denom;
  float vx = reduce(vx_p) / denom;
  float cdr[10]; float cdsum = 0.f;
  for (int c = 0; c < 10; c++) { cdr[c] = reduce(cd[c]); cdsum += cdr[c]; }
  if (t == 0) {
    float height = 2.f * sqrtf(vy + 1e-8f);
    float width  = 2.f * sqrtf(vx + 1e-8f);
    float* o = out + (size_t)(b * NS + s) * 17;
    o[0] = sumraw * (1.f / 900.f);
    o[1] = cy; o[2] = cx; o[3] = height; o[4] = width;
    o[5] = height * width; o[6] = height / (width + 1e-8f);
    float inv = 1.f / (cdsum + 1e-8f);
    for (int c = 0; c < 10; c++) o[7 + c] = cdr[c] * inv;
  }
}

// ---------------------------------------------------------------------------
extern "C" void kernel_launch(void* const* d_in, const int* in_sizes, int n_in,
                              void* d_out, int out_size, void* d_ws, size_t ws_size,
                              hipStream_t stream) {
  const int*   grid      = (const int*)d_in[0];
  const float* noise     = (const float*)d_in[1];
  const float* cemb      = (const float*)d_in[2];
  const float* pemb      = (const float*)d_in[3];
  const float* nin_g     = (const float*)d_in[4];
  const float* nin_b     = (const float*)d_in[5];
  const float* slot_mu   = (const float*)d_in[6];
  const float* slot_sg   = (const float*)d_in[7];
  const float* nsl_g     = (const float*)d_in[8];
  const float* nsl_b     = (const float*)d_in[9];
  const float* Wq        = (const float*)d_in[10];
  const float* Wk        = (const float*)d_in[11];
  const float* Wv        = (const float*)d_in[12];
  const float* gWih      = (const float*)d_in[13];
  const float* gWhh      = (const float*)d_in[14];
  const float* gbih      = (const float*)d_in[15];
  const float* gbhh      = (const float*)d_in[16];
  const float* W1        = (const float*)d_in[17];
  const float* b1        = (const float*)d_in[18];
  const float* W2        = (const float*)d_in[19];
  const float* b2        = (const float*)d_in[20];

  char* ws = (char*)d_ws;
  size_t off = 0;
  auto alloc = [&](size_t bytes) -> void* {
    off = (off + 255) & ~(size_t)255;
    void* p = ws + off;
    off += bytes;
    return p;
  };

  const size_t MROWS = (size_t)NB * NPAD;  // 475136 pixel rows (padded)
  _Float16* Wqt   = (_Float16*)alloc(128 * 128 * 2);
  _Float16* Wkt   = (_Float16*)alloc(128 * 128 * 2);
  _Float16* Wvt   = (_Float16*)alloc(128 * 128 * 2);
  _Float16* WihT  = (_Float16*)alloc(384 * 128 * 2);
  _Float16* WhhT  = (_Float16*)alloc(384 * 128 * 2);
  _Float16* W1t   = (_Float16*)alloc(256 * 128 * 2);
  _Float16* W2t   = (_Float16*)alloc(128 * 256 * 2);
  _Float16* Xh    = (_Float16*)alloc(MROWS * DDIM * 2);          // ~121.6 MB
  _Float16* Kh    = (_Float16*)alloc(MROWS * DDIM * 2);          // ~121.6 MB
  _Float16* VTh   = (_Float16*)alloc((size_t)NB * DDIM * NPAD * 2); // ~121.6 MB
  float*    slots = (float*)alloc((size_t)NB * SPAD * DDIM * 4);
  _Float16* slotsH= (_Float16*)alloc((size_t)NB * SPAD * DDIM * 2);
  _Float16* lnqH  = (_Float16*)alloc((size_t)NB * SPAD * DDIM * 2);
  _Float16* qH    = (_Float16*)alloc((size_t)NB * SPAD * DDIM * 2);
  float*    attn  = (float*)alloc((size_t)NB * SPAD * NPAD * 4); // ~30.4 MB
  _Float16* updH  = (_Float16*)alloc((size_t)NB * SPAD * DDIM * 2);
  float*    gi    = (float*)alloc((size_t)NB * SPAD * 384 * 4);
  float*    gh    = (float*)alloc((size_t)NB * SPAD * 384 * 4);
  _Float16* h1H   = (_Float16*)alloc((size_t)NB * SPAD * 256 * 2);
  float*    mlpo  = (float*)alloc((size_t)NB * SPAD * DDIM * 4);

  // --- weight conversion (transposed to [out,in] f16) ---
  wconv_kernel<<<(128 * 128 + 255) / 256, 256, 0, stream>>>(Wq, Wqt, 128, 128, 1);
  wconv_kernel<<<(128 * 128 + 255) / 256, 256, 0, stream>>>(Wk, Wkt, 128, 128, 1);
  wconv_kernel<<<(128 * 128 + 255) / 256, 256, 0, stream>>>(Wv, Wvt, 128, 128, 1);
  wconv_kernel<<<(384 * 128 + 255) / 256, 256, 0, stream>>>(gWih, WihT, 384, 128, 0);
  wconv_kernel<<<(384 * 128 + 255) / 256, 256, 0, stream>>>(gWhh, WhhT, 384, 128, 0);
  wconv_kernel<<<(256 * 128 + 255) / 256, 256, 0, stream>>>(W1, W1t, 256, 128, 1);
  wconv_kernel<<<(128 * 256 + 255) / 256, 256, 0, stream>>>(W2, W2t, 128, 256, 1);

  // --- embed + LN, slot init ---
  embed_ln_kernel<<<dim3((unsigned)(MROWS / 8)), dim3(32, 8), 0, stream>>>(
      grid, cemb, pemb, nin_g, nin_b, Xh);
  slots_init_kernel<<<(NB * SPAD * DDIM + 255) / 256, 256, 0, stream>>>(
      noise, slot_mu, slot_sg, slots, slotsH);

  // --- K/V projection (big WMMA GEMM, A-fragment reuse over 16 n-tiles) ---
  kv_gemm_kernel<<<dim3((unsigned)(MROWS / 16)), 32, 0, stream>>>(Xh, Wkt, Wvt, Kh, VTh);

  const int MT = NB * SPAD / 16;  // 512 M-tiles for slot-row GEMMs
  for (int it = 0; it < 3; it++) {
    // attend
    slot_ln_kernel<<<NB * SPAD / 8, dim3(32, 8), 0, stream>>>(slots, nsl_g, nsl_b, lnqH);
    gemm16_kernel<<<dim3(MT, 8), 32, 0, stream>>>(lnqH, DDIM, Wqt, DDIM,
                                                  nullptr, qH, DDIM, DDIM, nullptr, 0);
    logits_kernel<<<dim3(NB, 2), 32, 0, stream>>>(qH, Kh, attn);
    softmax_kernel<<<dim3(NB, (NPAD + 255) / 256), 256, 0, stream>>>(attn);
    renorm_kernel<<<NB * SPAD, 256, 0, stream>>>(attn);
    // updates = attn @ v
    updates_kernel<<<NB, 32, 0, stream>>>(attn, VTh, updH);
    // GRU
    gemm16_kernel<<<dim3(MT, 384 / 16), 32, 0, stream>>>(updH, DDIM, WihT, DDIM,
                                                         gi, nullptr, 384, DDIM, nullptr, 0);
    gemm16_kernel<<<dim3(MT, 384 / 16), 32, 0, stream>>>(slotsH, DDIM, WhhT, DDIM,
                                                         gh, nullptr, 384, DDIM, nullptr, 0);
    gru_kernel<<<(NB * NS * DDIM + 255) / 256, 256, 0, stream>>>(gi, gh, gbih, gbhh,
                                                                 slots, slotsH);
    // MLP residual
    gemm16_kernel<<<dim3(MT, 256 / 16), 32, 0, stream>>>(slotsH, DDIM, W1t, DDIM,
                                                         nullptr, h1H, 256, DDIM, b1, 1);
    gemm16_kernel<<<dim3(MT, DDIM / 16), 32, 0, stream>>>(h1H, 256, W2t, 256,
                                                          mlpo, nullptr, DDIM, 256, b2, 0);
    mlp_add_kernel<<<(NB * NS * DDIM + 255) / 256, 256, 0, stream>>>(mlpo, slots, slotsH);
  }

  // final attend
  slot_ln_kernel<<<NB * SPAD / 8, dim3(32, 8), 0, stream>>>(slots, nsl_g, nsl_b, lnqH);
  gemm16_kernel<<<dim3(MT, 8), 32, 0, stream>>>(lnqH, DDIM, Wqt, DDIM,
                                                nullptr, qH, DDIM, DDIM, nullptr, 0);
  logits_kernel<<<dim3(NB, 2), 32, 0, stream>>>(qH, Kh, attn);
  softmax_kernel<<<dim3(NB, (NPAD + 255) / 256), 256, 0, stream>>>(attn);
  renorm_kernel<<<NB * SPAD, 256, 0, stream>>>(attn);

  // outputs: [slots | attn | props]
  float* out       = (float*)d_out;
  float* out_attn  = out + (size_t)NB * NS * DDIM;
  float* out_props = out_attn + (size_t)NB * NS * NPIX;
  out_slots_kernel<<<(NB * NS * DDIM + 255) / 256, 256, 0, stream>>>(slots, out);
  out_attn_kernel<<<(NB * NS * NPIX + 255) / 256, 256, 0, stream>>>(attn, out_attn);
  props_kernel<<<NB * NS, 256, 0, stream>>>(attn, grid, out_props);
  (void)in_sizes; (void)n_in; (void)out_size; (void)ws_size;
}
